// Encoder_Decoder_11123965296670
// MI455X (gfx1250) — compile-verified, matrix operand order checked by reading, not run on hardware
//
#include <hip/hip_runtime.h>
#include <hip/hip_bf16.h>

// ---------------------------------------------------------------------------
// Encoder/decoder LSTM + attention for MI455X (gfx1250), wave32 + WMMA bf16.
//
// Sizes: V=50257, H=512, T=256, B=64.
// GEMMs use v_wmma_f32_16x16x32_bf16 with fp32 accumulation; operands are
// pre-packed into the exact ISA fragment layouts (CDNA5 ISA §7.12.2)
// so each lane loads its 16 bf16 elements as one contiguous 32-byte chunk.
// The LSTM-step K loop is hand modulo-scheduled with ping-pong fragment sets
// (unroll disabled) so exactly two fragment generations are live: spill-free,
// with one full load-group always in flight ahead of each WMMA group.
// The attention kernel stages its per-t 16 KB fragment slab into LDS with the
// Tensor Data Mover (tensor_load_to_lds, double-buffered, s_wait_tensorcnt).
// ---------------------------------------------------------------------------

typedef __attribute__((ext_vector_type(16))) __bf16 v16bf;
typedef __attribute__((ext_vector_type(8)))  float  v8f;
typedef __attribute__((ext_vector_type(4)))  unsigned int u32x4;
typedef __attribute__((ext_vector_type(8)))  int    i32x8;
typedef __attribute__((ext_vector_type(4)))  int    i32x4;

#define TT   256
#define BB   64
#define HH   512
#define VV   50257
#define LGS  50304          // padded logits row stride (mult of 16)

#define HAVE_TDM (__has_builtin(__builtin_amdgcn_tensor_load_to_lds) && \
                  __has_builtin(__builtin_amdgcn_s_wait_tensorcnt))

__device__ __forceinline__ v8f zero8() {
  v8f z = {0.f,0.f,0.f,0.f,0.f,0.f,0.f,0.f};
  return z;
}

__device__ __forceinline__ v8f wmma_bf16(v16bf a, v16bf b, v8f c) {
  // 8 args: (neg_a, A, neg_b, B, c_mod, C, reuse_a, reuse_b)
  return __builtin_amdgcn_wmma_f32_16x16x32_bf16(false, a, false, b,
                                                 (short)0, c, false, false);
}

__device__ __forceinline__ v16bf ldfrag(const unsigned short* p) {
  return *reinterpret_cast<const v16bf*>(p);
}

__device__ __forceinline__ unsigned short f2bf(float f) {
  unsigned int u = __float_as_uint(f);
  unsigned int r = (u + 0x7FFFu + ((u >> 16) & 1u)) >> 16;
  return (unsigned short)r;
}

__device__ __forceinline__ float sigm(float x) {
  return 1.f / (1.f + __expf(-x));
}

// 16-bit A/B fragment K-index mapping (ISA §7.12.2):
//   kk = (e&7) + 8*laneHi + 16*(e>>3)

// --- init: zero c-state and h0 pack ----------------------------------------
__global__ void init_state_kernel(float* __restrict__ C,
                                  unsigned short* __restrict__ H0) {
  int i = blockIdx.x * 256 + threadIdx.x;   // 32768
  C[i]  = 0.f;
  H0[i] = 0;
}

// --- pack combined LSTM weights into B-fragment layout ---------------------
// WB[kt(32)][gt(128)][lane(32)][e(16)]  (K = [x|h] cat dim, N = gate col)
__global__ void pack_w_kernel(const float* __restrict__ W_ih,
                              const float* __restrict__ W_hh,
                              unsigned short* __restrict__ WB) {
  int idx = blockIdx.x * 256 + threadIdx.x;          // 2^21 elements
  int e  = idx & 15;
  int ln = (idx >> 4) & 31;
  int gt = (idx >> 9) & 127;
  int kt = idx >> 16;
  int g  = gt * 16 + (ln & 15);
  int kk = (e & 7) + ((ln >> 4) & 1) * 8 + ((e >> 3) << 4);
  int k  = kt * 32 + kk;
  float v = (k < HH) ? W_ih[g * HH + k] : W_hh[g * HH + (k - HH)];
  WB[idx] = f2bf(v);
}

// --- gather embeddings, pack into A-fragment layout ------------------------
// XPack[t][mt(4)][kt(16)][lane(32)][e(16)] ; t==256 -> target embeddings
__global__ void pack_x_kernel(const int* __restrict__ input_lines,
                              const int* __restrict__ target_lines,
                              const float* __restrict__ emb_in,
                              const float* __restrict__ emb_tgt,
                              unsigned short* __restrict__ XPack,
                              unsigned short* __restrict__ TgtPack) {
  int idx = blockIdx.x * 256 + threadIdx.x;          // 257*32768 elements
  int e   = idx & 15;
  int ln  = (idx >> 4) & 31;
  int kt  = (idx >> 9) & 15;
  int mtl = (idx >> 13) & 3;
  int t   = idx >> 15;
  int b   = mtl * 16 + (ln & 15);
  int kk  = (e & 7) + ((ln >> 4) & 1) * 8 + ((e >> 3) << 4);
  int k   = kt * 32 + kk;
  if (t < TT) {
    int row = input_lines[t * BB + b];
    XPack[idx] = f2bf(emb_in[row * HH + k]);
  } else {
    int row = target_lines[b];                       // target_lines[0][b]
    TgtPack[idx & 32767] = f2bf(emb_tgt[row * HH + k]);
  }
}

// --- one LSTM step: gates = [x|h] @ Wcat.T (WMMA) -> LDS -> activations ----
// grid = 16 blocks (mt = blk/4, r = blk%4), 256 threads (8 waves).
// mode 0: encoder (write hf32 / hnext / hs_t=HSall[t]);
// mode 1: decoder (write hs_t = AttA, ht transposed A-fragment layout).
#define GST   514
#define WSTEP 65536                                  // 128 tiles * 512 elems

// load fragment set S for K-chunk KT (A from APTR)
#define LOADSET(S, APTR, KT)                                                 \
  { const unsigned short* wp_ = wbase + (size_t)(KT) * WSTEP;                \
    a##S    = ldfrag(APTR);                                                  \
    b##S##0 = ldfrag(wp_ + gt0 * 512); b##S##1 = ldfrag(wp_ + gt1 * 512);    \
    b##S##2 = ldfrag(wp_ + gt2 * 512); b##S##3 = ldfrag(wp_ + gt3 * 512); }

#define WMMASET(S)                                                           \
  acc0 = wmma_bf16(a##S, b##S##0, acc0);                                     \
  acc1 = wmma_bf16(a##S, b##S##1, acc1);                                     \
  acc2 = wmma_bf16(a##S, b##S##2, acc2);                                     \
  acc3 = wmma_bf16(a##S, b##S##3, acc3);

#define STORE_ACC(ACC, TTJ)                                                  \
  { int c0 = ((TTJ) >> 3) * 128 + ((TTJ) & 7) * 16 + (lane & 15);            \
    _Pragma("unroll")                                                        \
    for (int v = 0; v < 8; ++v) gbuf[(mrow + v) * GST + c0] = (ACC)[v]; }

__global__ void lstm_step_kernel(const unsigned short* __restrict__ xpack,
                                 const unsigned short* __restrict__ hprev,
                                 const unsigned short* __restrict__ wb,
                                 const float* __restrict__ b_ih,
                                 const float* __restrict__ b_hh,
                                 float* __restrict__ cstate,
                                 unsigned short* __restrict__ hnext,
                                 float* __restrict__ hf32,
                                 unsigned short* __restrict__ hs_t,
                                 int mode) {
  __shared__ float gbuf[16 * GST];                   // [16 b][512 gate cols]
  int wave = threadIdx.x >> 5, lane = threadIdx.x & 31;
  int mt = blockIdx.x >> 2, r = blockIdx.x & 3;
  int laneOff = lane * 16;

  v8f acc0 = zero8(), acc1 = zero8(), acc2 = zero8(), acc3 = zero8();
  int tt0 = wave * 4;
  int gt0 = ((tt0 + 0) >> 3) * 32 + r * 8 + ((tt0 + 0) & 7);
  int gt1 = ((tt0 + 1) >> 3) * 32 + r * 8 + ((tt0 + 1) & 7);
  int gt2 = ((tt0 + 2) >> 3) * 32 + r * 8 + ((tt0 + 2) & 7);
  int gt3 = ((tt0 + 3) >> 3) * 32 + r * 8 + ((tt0 + 3) & 7);

  const unsigned short* xbase = xpack + (mt * 16) * 512 + laneOff;
  const unsigned short* hbase = hprev + (mt * 16) * 512 + laneOff;
  const unsigned short* wbase = wb + laneOff;

  // Hand modulo-scheduled pipeline, period 2, ping-pong sets {0,1}:
  // exactly two fragment generations live at any time -> no spills, and one
  // full load-group is in flight ahead of every WMMA group.
  // K-chunk c: A fragment from x (c<16) or h (c>=16).
  v16bf a0, b00, b01, b02, b03;
  v16bf a1, b10, b11, b12, b13;

  LOADSET(0, xbase, 0);                              // chunk 0
#pragma clang loop unroll(disable)
  for (int kt = 0; kt < 30; kt += 2) {
    const unsigned short* ap1 =
        (kt + 1 < 16) ? (xbase + (kt + 1) * 512) : (hbase + (kt - 15) * 512);
    LOADSET(1, ap1, kt + 1);                         // chunk kt+1
    WMMASET(0);                                      // chunk kt
    const unsigned short* ap2 =
        (kt + 2 < 16) ? (xbase + (kt + 2) * 512) : (hbase + (kt - 14) * 512);
    LOADSET(0, ap2, kt + 2);                         // chunk kt+2
    WMMASET(1);                                      // chunk kt+1
  }
  LOADSET(1, hbase + 15 * 512, 31);                  // chunk 31
  WMMASET(0);                                        // chunk 30
  WMMASET(1);                                        // chunk 31

  // D layout: vgpr v -> M = v + 8*(lane/16), N = lane%16
  int mrow = (lane >> 4) * 8;
  STORE_ACC(acc0, tt0 + 0);
  STORE_ACC(acc1, tt0 + 1);
  STORE_ACC(acc2, tt0 + 2);
  STORE_ACC(acc3, tt0 + 3);
  __syncthreads();

  // activations: 16 b x 128 cols = 2048 elems, 8 per thread
#pragma unroll
  for (int it = 0; it < 8; ++it) {
    int eidx = it * 256 + threadIdx.x;
    int bl = eidx >> 7;                              // 0..15
    int lc = eidx & 127;
    int jj = r * 128 + lc;                           // 0..511 within gate
    int b  = mt * 16 + bl;
    float gi = gbuf[bl * GST + 0 * 128 + lc] + b_ih[jj]          + b_hh[jj];
    float gf = gbuf[bl * GST + 1 * 128 + lc] + b_ih[HH + jj]     + b_hh[HH + jj];
    float gg = gbuf[bl * GST + 2 * 128 + lc] + b_ih[2 * HH + jj] + b_hh[2 * HH + jj];
    float go = gbuf[bl * GST + 3 * 128 + lc] + b_ih[3 * HH + jj] + b_hh[3 * HH + jj];
    float i_ = sigm(gi), f_ = sigm(gf), o_ = sigm(go);
    float g_ = tanhf(gg);
    float cold = cstate[b * HH + jj];
    float cnew = f_ * cold + i_ * g_;
    cstate[b * HH + jj] = cnew;
    float h = o_ * tanhf(cnew);
    unsigned short hb = f2bf(h);
    if (mode == 0) {
      hf32[b * HH + jj] = h;
      // next-step A pack: m=b, k=jj
      { int kt = jj >> 5, kk = jj & 31;
        int ln = (b & 15) + (((kk >> 3) & 1) << 4);
        int e  = (kk & 7) + ((kk >> 4) << 3);
        hnext[((mt * 16 + kt) * 32 + ln) * 16 + e] = hb; }
      // attention B pack (hs_all[t]): k=b, n=jj
      { int et = jj >> 4, kc = b >> 5, kk = b & 31;
        int ln = (jj & 15) + (((kk >> 3) & 1) << 4);
        int e  = (kk & 7) + ((kk >> 4) << 3);
        hs_t[((et * 2 + kc) * 32 + ln) * 16 + e] = hb; }
    } else {
      // attention A pack (ht transposed): m=jj, k=b
      int kc = b >> 5, kk = b & 31;
      int ln = (jj & 15) + (((kk >> 3) & 1) << 4);
      int e  = (kk & 7) + ((kk >> 4) << 3);
      hs_t[(((jj >> 4) * 2 + kc) * 32 + ln) * 16 + e] = hb;
    }
  }
}

// --- vocab projection: logits = hs @ W_out.T + b_out (bandwidth-bound) -----
#define HKS 132
__global__ void logits_kernel(const float* __restrict__ hf32,
                              const float* __restrict__ W_out,
                              const float* __restrict__ b_out,
                              float* __restrict__ Lg) {
  __shared__ float hsl[64 * HKS];                    // 64 b x 128 k chunk
  int tid = threadIdx.x;
  int v  = blockIdx.x * 64 + (tid >> 2);
  int b0 = (tid & 3) * 16;
  float acc[16];
#pragma unroll
  for (int i = 0; i < 16; ++i) acc[i] = 0.f;
  for (int kb = 0; kb < 4; ++kb) {
    __syncthreads();
    for (int i = tid; i < 64 * 128; i += 256) {
      int bb = i >> 7, kk = i & 127;
      hsl[bb * HKS + kk] = hf32[bb * HH + kb * 128 + kk];
    }
    __syncthreads();
    if (v < VV) {
      const float* wrow = W_out + v * HH + kb * 128;
      for (int k = 0; k < 128; ++k) {
        float w = wrow[k];
#pragma unroll
        for (int i = 0; i < 16; ++i)
          acc[i] = fmaf(w, hsl[(b0 + i) * HKS + k], acc[i]);
      }
    }
  }
  if (v < VV) {
    float bo = b_out[v];
#pragma unroll
    for (int i = 0; i < 16; ++i)
      Lg[(b0 + i) * LGS + v] = acc[i] + bo;
  }
}

// --- per-batch NLL contribution (max / sumexp reduction) -------------------
__global__ void loss_partial_kernel(const float* __restrict__ Lg,
                                    const int* __restrict__ target_lines,
                                    float* __restrict__ Lpart) {
  __shared__ float red[256];
  int b = blockIdx.x, tid = threadIdx.x;
  const float* row = Lg + b * LGS;
  float mx = -3.4e38f;
  for (int v = tid; v < VV; v += 256) mx = fmaxf(mx, row[v]);
  red[tid] = mx; __syncthreads();
  for (int s = 128; s > 0; s >>= 1) {
    if (tid < s) red[tid] = fmaxf(red[tid], red[tid + s]);
    __syncthreads();
  }
  mx = red[0]; __syncthreads();
  float sum = 0.f;
  for (int v = tid; v < VV; v += 256) sum += __expf(row[v] - mx);
  red[tid] = sum; __syncthreads();
  for (int s = 128; s > 0; s >>= 1) {
    if (tid < s) red[tid] += red[tid + s];
    __syncthreads();
  }
  if (tid == 0) {
    int tgt = target_lines[b];
    float lp = row[tgt] - mx - __logf(red[0]);
    Lpart[b] = -lp * (1.0f / (float)BB);
  }
}

__global__ void loss_final_kernel(const float* __restrict__ Lpart,
                                  float* __restrict__ out) {
  __shared__ float red[64];
  int tid = threadIdx.x;
  red[tid] = Lpart[tid]; __syncthreads();
  for (int s = 32; s > 0; s >>= 1) {
    if (tid < s) red[tid] += red[tid + s];
    __syncthreads();
  }
  if (tid == 0) out[0] = red[0];
}

#if HAVE_TDM
// TDM: 1-D copy of one 16 KB fragment slab (2048 x 8-byte elements) to LDS.
// D# per CDNA5 ISA: group0 = {count, lds_addr, global_addr,
// type=2}; group1 = {data_size=3, tensor_dim0=2048, tensor_dim1=1,
// tile_dim0=2048, tile_dim1=1, tensor_dim0_stride=2048}; groups 2/3 unused.
__device__ __forceinline__ void tdm_load_16k(const unsigned short* gsrc,
                                             unsigned lds_off) {
  unsigned long long ga = (unsigned long long)(size_t)gsrc;
  u32x4 g0 = { 1u,                                   // count=1, user mode
               lds_off,                              // lds_addr (bytes)
               (unsigned)(ga & 0xFFFFFFFFu),         // global_addr[31:0]
               (unsigned)((ga >> 32) & 0x01FFFFFFu)  // global_addr[56:32]
                   | (2u << 30) };                   // type=2 ("image")
  i32x8 g1 = { (int)0x00030000,        // wg_mask=0, data_size=3 (8B)
               (int)(2048u << 16),     // tensor_dim0[15:0] @ bits 63:48
               (int)(1u << 16),        // tensor_dim0 hi=0 | tensor_dim1 lo=1
               (int)(2048u << 16),     // tensor_dim1 hi=0 | tile_dim0=2048
               1,                      // tile_dim1=1, tile_dim2=0
               2048,                   // tensor_dim0_stride lo32
               0, 0 };
  i32x4 z4 = {0, 0, 0, 0};
#if __clang_major__ >= 23
  i32x8 z8 = {0, 0, 0, 0, 0, 0, 0, 0};
  __builtin_amdgcn_tensor_load_to_lds(g0, g1, z4, z4, z8, 0);
#else
  __builtin_amdgcn_tensor_load_to_lds(g0, g1, z4, z4, 0);
#endif
}
#endif

// --- attention: at[h,e] = exp(S_0)/sum_t exp(S_t), S_t = ht.T @ hs_all[t] --
// 1024 tiles (32 mta x 32 et); block = 8 waves covering 8 consecutive et of
// one mta, so its per-t working set is one contiguous 16 KB slab of HSall.
__global__ void attention_kernel(const unsigned short* __restrict__ AttA,
                                 const unsigned short* __restrict__ HSall,
                                 float* __restrict__ at_out) {
  int wave = threadIdx.x >> 5, lane = threadIdx.x & 31;
  int tile = blockIdx.x * 8 + wave;
  int mta = tile >> 5, et = tile & 31;
  v16bf a0 = ldfrag(AttA + ((mta * 2 + 0) * 32 + lane) * 16);
  v16bf a1 = ldfrag(AttA + ((mta * 2 + 1) * 32 + lane) * 16);
  float sum[8], e0[8];
#pragma unroll
  for (int v = 0; v < 8; ++v) { sum[v] = 0.f; e0[v] = 0.f; }

#if HAVE_TDM
  __shared__ __align__(32) unsigned short abuf[2][8192];   // 2 x 16 KB slabs
  int et0 = (blockIdx.x & 3) * 8;                          // block's et base
  const unsigned short* slab0 = HSall + (size_t)(0 * 32 + et0) * 1024;
  if (wave == 0) tdm_load_16k(slab0, (unsigned)(size_t)(&abuf[0][0]));
  for (int t = 0; t < TT; ++t) {
    if (wave == 0) {
      if (t + 1 < TT) {
        tdm_load_16k(HSall + (size_t)((t + 1) * 32 + et0) * 1024,
                     (unsigned)(size_t)(&abuf[(t + 1) & 1][0]));
        __builtin_amdgcn_s_wait_tensorcnt(1);   // slab t complete (in-order)
      } else {
        __builtin_amdgcn_s_wait_tensorcnt(0);
      }
    }
    __syncthreads();                            // slab t visible to all waves
    const unsigned short* bp = &abuf[t & 1][wave * 1024 + lane * 16];
    v16bf b0 = ldfrag(bp);
    v16bf b1 = ldfrag(bp + 512);
    v8f acc = zero8();
    acc = wmma_bf16(a0, b0, acc);
    acc = wmma_bf16(a1, b1, acc);
#pragma unroll
    for (int v = 0; v < 8; ++v) {
      float ev = __expf(acc[v]);
      sum[v] += ev;
      if (t == 0) e0[v] = ev;
    }
    __syncthreads();                  // all reads of slab t done before reuse
  }
#else
  for (int t = 0; t < TT; ++t) {
    const unsigned short* bp = HSall + ((t * 32 + et) * 2) * 512 + lane * 16;
    if (t + 1 < TT)
      __builtin_prefetch(bp + 32 * 2 * 512, 0, 0);   // next-t fragment
    v16bf b0 = ldfrag(bp);
    v16bf b1 = ldfrag(bp + 512);
    v8f acc = zero8();
    acc = wmma_bf16(a0, b0, acc);
    acc = wmma_bf16(a1, b1, acc);
#pragma unroll
    for (int v = 0; v < 8; ++v) {
      float ev = __expf(acc[v]);
      sum[v] += ev;
      if (t == 0) e0[v] = ev;
    }
  }
#endif

  int ncol  = et * 16 + (lane & 15);
  int mbase = mta * 16 + (lane >> 4) * 8;
#pragma unroll
  for (int v = 0; v < 8; ++v)
    at_out[(mbase + v) * HH + ncol] = e0[v] / sum[v];
}

// ---------------------------------------------------------------------------
extern "C" void kernel_launch(void* const* d_in, const int* in_sizes, int n_in,
                              void* d_out, int out_size, void* d_ws, size_t ws_size,
                              hipStream_t stream) {
  (void)in_sizes; (void)n_in; (void)out_size; (void)ws_size;
  const int*   input_lines  = (const int*)d_in[0];
  const int*   target_lines = (const int*)d_in[1];
  const float* emb_in  = (const float*)d_in[2];
  const float* emb_tgt = (const float*)d_in[3];
  const float* W_ih    = (const float*)d_in[4];
  const float* W_hh    = (const float*)d_in[5];
  const float* b_ih    = (const float*)d_in[6];
  const float* b_hh    = (const float*)d_in[7];
  const float* W_out   = (const float*)d_in[8];
  const float* b_out   = (const float*)d_in[9];
  float* out = (float*)d_out;

  // workspace layout (bytes), ~51.2 MB total
  char* ws = (char*)d_ws;
  unsigned short* WB    = (unsigned short*)(ws + 0);          //  4,194,304
  unsigned short* XP    = (unsigned short*)(ws + 4194304);    // 16,777,216
  unsigned short* TgtP  = (unsigned short*)(ws + 20971520);   //     65,536
  unsigned short* HA    = (unsigned short*)(ws + 21037056);   //     65,536
  unsigned short* HB    = (unsigned short*)(ws + 21102592);   //     65,536
  unsigned short* HSall = (unsigned short*)(ws + 21168128);   // 16,777,216
  unsigned short* AttA  = (unsigned short*)(ws + 37945344);   //     65,536
  float*          Cbuf  = (float*)(ws + 38010880);            //    131,072
  float*          Hf32  = (float*)(ws + 38141952);            //    131,072
  float*          Lg    = (float*)(ws + 38273024);            // 12,877,824
  float*          Lpart = (float*)(ws + 51150848);            //        256

  // 1. zero c0 / h0
  init_state_kernel<<<128, 256, 0, stream>>>(Cbuf, HA);
  // 2. pack weights (bf16, B-fragment layout)
  pack_w_kernel<<<8192, 256, 0, stream>>>(W_ih, W_hh, WB);
  // 3. gather + pack embeddings (encoder + target)
  pack_x_kernel<<<32896, 256, 0, stream>>>(input_lines, target_lines,
                                           emb_in, emb_tgt, XP, TgtP);
  // 4. encoder recurrence (sequential, double-buffered h pack)
  for (int t = 0; t < TT; ++t) {
    const unsigned short* hprev = (t & 1) ? HB : HA;
    unsigned short*       hnext = (t & 1) ? HA : HB;
    lstm_step_kernel<<<16, 256, 0, stream>>>(
        XP + (size_t)t * 32768, hprev, WB, b_ih, b_hh,
        Cbuf, hnext, Hf32, HSall + (size_t)t * 32768, 0);
  }
  // 5. vocab projection from final hs (hf32), bandwidth-bound
  logits_kernel<<<786, 256, 0, stream>>>(Hf32, W_out, b_out, Lg);
  // 6. loss reduction
  loss_partial_kernel<<<64, 256, 0, stream>>>(Lg, target_lines, Lpart);
  loss_final_kernel<<<1, 64, 0, stream>>>(Lpart, out);
  // 7. decoder LSTM step: x = target emb, h = hs (in HA after t=255), c = cx
  lstm_step_kernel<<<16, 256, 0, stream>>>(
      TgtP, HA, WB, b_ih, b_hh, Cbuf, HB, Hf32, AttA, 1);
  // 8. attention: at = exp(S_0) / sum_t exp(S_t)  -> out[1..262144]
  attention_kernel<<<128, 256, 0, stream>>>(AttA, HSall, out + 1);
}
